// model_51659866637003
// MI455X (gfx1250) — matrix operand order repaired and checked by reading, not compiled, not run on hardware
//
#include <hip/hip_runtime.h>

// Problem sizes (fixed by the reference)
#define Bsz 1024
#define Ksz 1024
#define Dsz 500
#define Dp  512   // D padded to a multiple of 32 (bf16 WMMA K-step)
#define Hsz 128

typedef __attribute__((ext_vector_type(16))) __bf16       v16bf;
typedef __attribute__((ext_vector_type(8)))  float        v8f;
typedef __attribute__((ext_vector_type(4)))  unsigned int u32x4;
typedef __attribute__((ext_vector_type(8)))  unsigned int u32x8;

union Frag {
  v16bf bf;
  u32x8 u8;
  u32x4 u4[2];
};

__device__ __forceinline__ unsigned short f2bf(float f) {
  unsigned int u = __builtin_bit_cast(unsigned int, f);
  u += 0x7FFFu + ((u >> 16) & 1u);          // round-to-nearest-even
  return (unsigned short)(u >> 16);
}

// Tensor Data Mover: 1-D DMA of n_qwords * 8 bytes from global to LDS.
// D# packing per CDNA5 ISA 8.3/8.4:
//   group0: [1:0]count=1, [63:32]lds_addr, [120:64]global_addr, [127:126]type=2
//   group1: [17:16]data_size=3 (8B), [79:48]tensor_dim0, [127:112]tile_dim0,
//           [207:160]tensor_dim0_stride; mask/pad/iterate/barrier all 0.
__device__ __forceinline__ void tdm_load_to_lds(unsigned lds_addr,
                                                const void* gptr,
                                                unsigned n_qwords) {
  unsigned long long ga = (unsigned long long)gptr;
  u32x4 g0;
  g0.x = 1u;                                            // count=1
  g0.y = lds_addr;                                      // LDS byte address
  g0.z = (unsigned)(ga & 0xFFFFFFFFu);                  // global_addr lo
  g0.w = (unsigned)((ga >> 32) & 0x01FFFFFFu) | (2u << 30);  // ga hi | type=2
  u32x8 g1;
  g1.s0 = (3u << 16);                                   // data_size = 8 bytes
  g1.s1 = (n_qwords & 0xFFFFu) << 16;                   // tensor_dim0[15:0]
  g1.s2 = (n_qwords >> 16) & 0xFFFFu;                   // tensor_dim0[31:16]
  g1.s3 = (n_qwords & 0xFFFFu) << 16;                   // tile_dim0
  g1.s4 = 0u;                                           // tile_dim1/2 unused
  g1.s5 = n_qwords;                                     // dim0_stride lo32
  g1.s6 = 0u;
  g1.s7 = 0u;
  asm volatile("tensor_load_to_lds %0, %1" :: "s"(g0), "s"(g1) : "memory");
}

// Convert one row of a [rows, 500] f32 matrix to bf16 padded to 512, and
// compute sum(|row|) for the normalizer.
__global__ __launch_bounds__(256) void prep_rows(const float* __restrict__ src,
                                                 unsigned short* __restrict__ dst,
                                                 float* __restrict__ sums) {
  int row = blockIdx.x, t = threadIdx.x;
  float part = 0.f;
  for (int d = t; d < Dp; d += 256) {
    float v = (d < Dsz) ? src[row * Dsz + d] : 0.f;
    dst[row * Dp + d] = f2bf(v);
    part += __builtin_fabsf(v);
  }
  for (int m = 1; m < 32; m <<= 1) part += __shfl_xor(part, m, 32);
  __shared__ float red[8];
  if ((t & 31) == 0) red[t >> 5] = part;
  __syncthreads();
  if (t == 0) {
    float s = 0.f;
    for (int i = 0; i < 8; i++) s += red[i];
    sums[row] = s;
  }
}

// W0 [500,128] f32 -> W0T [128,512] bf16 (row h = column h of W0, zero-padded)
__global__ __launch_bounds__(256) void prep_w0t(const float* __restrict__ W0,
                                                unsigned short* __restrict__ w0t) {
  int h = blockIdx.x, t = threadIdx.x;
  for (int d = t; d < Dp; d += 256) {
    float v = (d < Dsz) ? W0[d * Hsz + h] : 0.f;
    w0t[h * Dp + d] = f2bf(v);
  }
}

// Main fused kernel.
// Grid: (K/16, B/8).  Block: 256 threads = 8 waves.
// Wave w: bgroup g = w>>1 owns 2 b's; parity p = w&1 owns H-columns [p*64, p*64+64).
// Each wave: acc[2 Mtiles][4 Ntiles], 16 D-chunks of 32 -> 128 WMMAs.
__global__ __launch_bounds__(256) void pair_mlp_wmma(
    const unsigned short* __restrict__ xb,
    const unsigned short* __restrict__ x1b,
    const unsigned short* __restrict__ w0t,
    const float* __restrict__ b0g,
    const float* __restrict__ w1g,
    const float* __restrict__ b1g,
    const float* __restrict__ sax,
    const float* __restrict__ sax1,
    float* __restrict__ out) {
  __shared__ unsigned short w0s[Hsz * Dp];   // 128 KB: all of W0T in LDS
  __shared__ unsigned short x1s[16 * Dp];    // 16 KB: x1 tile (16 k-rows)
  __shared__ float sbuf[2 * 8 * 16];         // 1 KB: cross-wave layer-2 partials

  const int tid   = threadIdx.x;
  const int k0    = blockIdx.x * 16;
  const int bbase = blockIdx.y * 8;
  const int wv    = tid >> 5, lane = tid & 31;

  // Stage W0T + x1 tile into LDS via the Tensor Data Mover (one wave issues;
  // TDM ignores EXEC and runs as an async DMA tracked by TENSORcnt).
  if (wv == 0) {
    tdm_load_to_lds((unsigned)(unsigned long long)(void*)w0s,
                    w0t, (Hsz * Dp * 2) / 8);
    tdm_load_to_lds((unsigned)(unsigned long long)(void*)x1s,
                    x1b + (size_t)k0 * Dp, (16 * Dp * 2) / 8);
    __builtin_amdgcn_s_wait_tensorcnt(0);
  }
  __syncthreads();

  const int g   = wv >> 1,  p    = wv & 1;
  const int hl  = lane >> 4, l15 = lane & 15;   // half-wave, lane-in-half
  const int nbase = p * 64;
  const int bj0   = bbase + g * 2;

  // A fragment (16-bit A 16x32): lane needs K = half*8+[0,8) and 16+half*8+[0,8)
  // -> two contiguous 16B runs per lane per chunk.  Row M = k-offset = l15.
  const char* x1base = (const char*)x1s + l15 * (Dp * 2) + hl * 16;
  const char* xbase0 = (const char*)xb + (size_t)bj0 * (Dp * 2) + hl * 16;
  const char* xbase1 = xbase0 + (Dp * 2);
  // B fragment (bf16 B 32x16): lane needs K = half*8+[0,8) and 16+half*8+[0,8)
  // of col N=l15 -> two contiguous 16B runs per lane from W0T row.
  const char* wcol[4];
#pragma unroll
  for (int t = 0; t < 4; t++)
    wcol[t] = (const char*)w0s + (nbase + t * 16 + l15) * (Dp * 2) + hl * 16;

  v8f acc[2][4];
#pragma unroll
  for (int j = 0; j < 2; j++)
#pragma unroll
    for (int t = 0; t < 4; t++) acc[j][t] = (v8f)(0.f);

  for (int d0b = 0; d0b < Dp * 2; d0b += 64) {     // 32 elements * 2B per chunk
    Frag x1f;
    x1f.u4[0] = *(const u32x4*)(x1base + d0b);
    x1f.u4[1] = *(const u32x4*)(x1base + d0b + 32);
    Frag a[2];
#pragma unroll
    for (int j = 0; j < 2; j++) {
      Frag xf;
      const char* xbj = j ? xbase1 : xbase0;
      xf.u4[0] = *(const u32x4*)(xbj + d0b);
      xf.u4[1] = *(const u32x4*)(xbj + d0b + 32);
      v16bf dv = xf.bf - x1f.bf;                    // packed bf16 subtract
      a[j].u8  = __builtin_bit_cast(u32x8, dv) & 0x7FFF7FFFu;  // |.| via bit-and
    }
#pragma unroll
    for (int t = 0; t < 4; t++) {
      Frag wfr;
      wfr.u4[0] = *(const u32x4*)(wcol[t] + d0b);
      wfr.u4[1] = *(const u32x4*)(wcol[t] + d0b + 32);
#pragma unroll
      for (int j = 0; j < 2; j++) {
        acc[j][t] = __builtin_amdgcn_wmma_f32_16x16x32_bf16(
            false, a[j].bf, false, wfr.bf, (short)0, acc[j][t], false, false);
      }
    }
  }

  // Fused layer 2: s = sum_n relu(h + b0[n]) * W1[n], reduced over the wave's
  // 64 N-columns (16 lanes x 4 tiles), then combined across the wave pair.
  float bb[4], ww[4];
#pragma unroll
  for (int t = 0; t < 4; t++) {
    int n = nbase + t * 16 + l15;
    bb[t] = b0g[n];
    ww[t] = w1g[n];
  }
#pragma unroll
  for (int j = 0; j < 2; j++) {
#pragma unroll
    for (int r = 0; r < 8; r++) {
      float s = 0.f;
#pragma unroll
      for (int t = 0; t < 4; t++) {
        float h = acc[j][t][r] + bb[t];
        h = fmaxf(h, 0.f);
        s = __builtin_fmaf(h, ww[t], s);
      }
      // reduce across the 16 lanes of this half (the N dimension)
      for (int m = 1; m < 16; m <<= 1) s += __shfl_xor(s, m, 32);
      if (l15 == 0)
        sbuf[p * 128 + (g * 2 + j) * 16 + (r + 8 * hl)] = s;
    }
  }
  __syncthreads();

  if (tid < 128) {
    int bl = tid >> 4, kk = tid & 15;
    float v = sbuf[bl * 16 + kk] + sbuf[128 + bl * 16 + kk];
    v += b1g[0];
    v = fmaxf(v, 0.f);
    int b = bbase + bl, k = k0 + kk;
    out[(size_t)b * Ksz + k] = v / (sax[b] + sax1[k]);
  }
}

extern "C" void kernel_launch(void* const* d_in, const int* in_sizes, int n_in,
                              void* d_out, int out_size, void* d_ws, size_t ws_size,
                              hipStream_t stream) {
  const float* x  = (const float*)d_in[0];
  const float* x1 = (const float*)d_in[1];
  const float* W0 = (const float*)d_in[2];
  const float* b0 = (const float*)d_in[3];
  const float* W1 = (const float*)d_in[4];
  const float* b1 = (const float*)d_in[5];
  float* out = (float*)d_out;

  // Workspace layout (needs ~2.14 MB):
  char* ws = (char*)d_ws;
  unsigned short* xb  = (unsigned short*)(ws);             // 1024*512*2 = 1 MB
  unsigned short* x1b = (unsigned short*)(ws + 1048576);   // 1 MB
  unsigned short* w0t = (unsigned short*)(ws + 2097152);   // 128 KB
  float* sax  = (float*)(ws + 2228224);                    // 4 KB
  float* sax1 = (float*)(ws + 2232320);                    // 4 KB

  prep_rows<<<Bsz, 256, 0, stream>>>(x, xb, sax);
  prep_rows<<<Ksz, 256, 0, stream>>>(x1, x1b, sax1);
  prep_w0t<<<Hsz, 256, 0, stream>>>(W0, w0t);

  dim3 grid(Ksz / 16, Bsz / 8);
  pair_mlp_wmma<<<grid, 256, 0, stream>>>(xb, x1b, w0t, b0, W1, b1,
                                          sax, sax1, out);
}
